// GATEmbedding_35098472743570
// MI455X (gfx1250) — matrix operand order repaired and checked by reading, not compiled, not run on hardware
//
#include <hip/hip_runtime.h>

// MI455X / gfx1250: wave32, WMMA 16x16x32 f16 -> f32 accumulate,
// async global->LDS DMA (ASYNCcnt) for double-buffered operand staging.
typedef __attribute__((ext_vector_type(16))) _Float16 v16h;
typedef __attribute__((ext_vector_type(8)))  _Float16 v8h;
typedef __attribute__((ext_vector_type(8)))  float    v8f;
typedef __attribute__((ext_vector_type(4)))  float    v4f;
typedef __attribute__((ext_vector_type(4)))  int      i32x4;

namespace {
constexpr int Bsz = 32;   // batch
constexpr int S   = 128;  // in channels (= out dim of final transpose)
constexpr int Nn  = 1024; // nodes
constexpr int Hh  = 2;    // heads
constexpr int Cc  = 128;  // out channels
constexpr float NEG = 0.2f;
constexpr int PAD = 40;   // halves per LDS row: 32 data + 8 pad (80B, 16B aligned)
}

// ---- gfx1250 async global->LDS copy (ASYNCcnt) with safe fallback ----------
#if defined(__has_builtin)
#if __has_builtin(__builtin_amdgcn_global_load_async_to_lds_b128) && \
    __has_builtin(__builtin_amdgcn_global_load_async_to_lds_b32)  && \
    __has_builtin(__builtin_amdgcn_s_wait_asynccnt)
#define HAVE_ASYNC 1
#endif
#endif
#ifndef HAVE_ASYNC
#define HAVE_ASYNC 0
#endif

#if HAVE_ASYNC
// addrspace-qualified pointer typedefs; casts go generic->ASn void* (real
// addrspacecast) then bitcast within the AS to the builtin's element type.
typedef __attribute__((address_space(1))) void*  as1vp;
typedef __attribute__((address_space(3))) void*  as3vp;
typedef __attribute__((address_space(1))) int*   as1ip;
typedef __attribute__((address_space(3))) int*   as3ip;
typedef __attribute__((address_space(1))) i32x4* as1qp;
typedef __attribute__((address_space(3))) i32x4* as3qp;
#define ASYNC_B128(g, l)                                      \
  __builtin_amdgcn_global_load_async_to_lds_b128(             \
      (as1qp)(as1vp)(g), (as3qp)(as3vp)(l), 0, 0)
#define ASYNC_B32(g, l)                                       \
  __builtin_amdgcn_global_load_async_to_lds_b32(              \
      (as1ip)(as1vp)(g), (as3ip)(as3vp)(l), 0, 0)
#define ASYNC_WAIT() __builtin_amdgcn_s_wait_asynccnt(0)
#else
#define ASYNC_WAIT() ((void)0)
#endif

__device__ __forceinline__ v16h make16(v8h lo, v8h hi) {
  return __builtin_shufflevector(lo, hi, 0,1,2,3,4,5,6,7,8,9,10,11,12,13,14,15);
}
__device__ __forceinline__ v8f zero8() {
  v8f z = {0.f,0.f,0.f,0.f,0.f,0.f,0.f,0.f};
  return z;
}

// ---------------------------------------------------------------------------
// Kernel 1: z = xt @ W per (b,h); emit zT[b,h,c,n] (f16) + s_src/s_dst (f32).
// Block = 128 threads (4 waves). Wave w owns n-rows [n0+16w, n0+16w+16),
// full C = 128 (8 WMMA tiles). K loop over S in chunks of 32.
// (Staging needs f32->f16 conversion, so raw-byte async DMA does not apply.)
// ---------------------------------------------------------------------------
__global__ __launch_bounds__(128) void gat_project(
    const float* __restrict__ x,        // [B,S,N]
    const float* __restrict__ Wm,       // [H,S,C]
    const float* __restrict__ att_src,  // [H,C]
    const float* __restrict__ att_dst,  // [H,C]
    _Float16* __restrict__ zT,          // [B,H,C,N] f16
    float* __restrict__ s_src,          // [B,H,N]
    float* __restrict__ s_dst)          // [B,H,N]
{
  __shared__ __align__(16) _Float16 ldsB[Cc * PAD];      // Bt[c][k]
  __shared__ __align__(16) _Float16 ldsA[4 * 16 * PAD];  // per-wave A[m][k]

  const int bh   = blockIdx.x;
  const int h    = bh & (Hh - 1);
  const int b    = bh >> 1;
  const int n0   = blockIdx.y * 64;
  const int t    = threadIdx.x;
  const int wave = t >> 5;
  const int lane = t & 31;
  const int lrow = lane & 15;
  const int hiH  = lane >> 4;
  const int nW   = n0 + wave * 16;

  v8f acc[8];
#pragma unroll
  for (int i = 0; i < 8; ++i) acc[i] = zero8();

  for (int s0 = 0; s0 < S; s0 += 32) {
    // Stage B tile: Bt[c=t][k=e] = W[h][s0+e][t]  (coalesced global reads)
#pragma unroll 4
    for (int e = 0; e < 32; ++e)
      ldsB[t * PAD + e] = (_Float16)Wm[((size_t)h * S + (s0 + e)) * Cc + t];
    // Stage A tile (per wave): A[m][k=lane] = x[b][s0+lane][nW+m]
    {
      const float* xp = x + ((size_t)b * S + (s0 + lane)) * Nn + nW;
      _Float16* ap = ldsA + (wave * 16) * PAD + lane;
#pragma unroll
      for (int q = 0; q < 4; ++q) {
        v4f x4 = *(const v4f*)(xp + 4 * q);
        ap[(4 * q + 0) * PAD] = (_Float16)x4.x;
        ap[(4 * q + 1) * PAD] = (_Float16)x4.y;
        ap[(4 * q + 2) * PAD] = (_Float16)x4.z;
        ap[(4 * q + 3) * PAD] = (_Float16)x4.w;
      }
    }
    __syncthreads();

    // A fragment: lane<16 -> K in {0..7, 16..23}; lane>=16 -> {8..15, 24..31}
    const _Float16* arow = ldsA + (wave * 16 + lrow) * PAD + hiH * 8;
    v16h afrag = make16(*(const v8h*)arow, *(const v8h*)(arow + 16));
#pragma unroll
    for (int tc = 0; tc < 8; ++tc) {
      // B fragment: lane<16 -> column N=lane, K in {0..7, 16..23}; lane>=16 -> {8..15, 24..31}
      const _Float16* brow = ldsB + (tc * 16 + lrow) * PAD + hiH * 8;
      v16h bfrag = make16(*(const v8h*)brow, *(const v8h*)(brow + 16));
      acc[tc] = __builtin_amdgcn_wmma_f32_16x16x32_f16(
          false, afrag, false, bfrag, (short)0, acc[tc], false, false);
    }
    __syncthreads();
  }

  // Attention score dot products (f32 accumulators for precision).
  float ps[8], pd[8];
#pragma unroll
  for (int r = 0; r < 8; ++r) { ps[r] = 0.f; pd[r] = 0.f; }
#pragma unroll
  for (int tc = 0; tc < 8; ++tc) {
    const int c = tc * 16 + lrow;
    const float as = att_src[h * Cc + c];
    const float ad = att_dst[h * Cc + c];
#pragma unroll
    for (int r = 0; r < 8; ++r) {
      ps[r] += acc[tc][r] * as;
      pd[r] += acc[tc][r] * ad;
    }
  }
#pragma unroll
  for (int mask = 1; mask < 16; mask <<= 1) {
#pragma unroll
    for (int r = 0; r < 8; ++r) {
      ps[r] += __shfl_xor(ps[r], mask, 32);
      pd[r] += __shfl_xor(pd[r], mask, 32);
    }
  }
  if (lrow < 8) {
    const int n = nW + hiH * 8 + lrow;
    s_src[(size_t)bh * Nn + n] = ps[lrow];
    s_dst[(size_t)bh * Nn + n] = pd[lrow];
  }

  // Store zT[b,h,c,n]: per (lane,tile) the 8 accumulator rows are 8
  // consecutive n for one c -> one 16B packed store.
#pragma unroll
  for (int tc = 0; tc < 8; ++tc) {
    const int c = tc * 16 + lrow;
    v8h zz;
#pragma unroll
    for (int r = 0; r < 8; ++r) zz[r] = (_Float16)acc[tc][r];
    *(v8h*)(zT + ((size_t)bh * Cc + c) * Nn + nW + hiH * 8) = zz;
  }
}

// ---------------------------------------------------------------------------
// Kernel 2: per-(b,h) global max of s_src (exact softmax max via monotone
// leaky-relu: rowmax_i = lrelu(d_i + max_j s_j)).
// ---------------------------------------------------------------------------
__global__ __launch_bounds__(128) void gat_smax(const float* __restrict__ s_src,
                                                float* __restrict__ smax) {
  __shared__ float red[128];
  const int bh = blockIdx.x, t = threadIdx.x;
  float m = -3.4e38f;
  for (int i = t; i < Nn; i += 128) m = fmaxf(m, s_src[(size_t)bh * Nn + i]);
  red[t] = m;
  __syncthreads();
  for (int s = 64; s > 0; s >>= 1) {
    if (t < s) red[t] = fmaxf(red[t], red[t + s]);
    __syncthreads();
  }
  if (t == 0) smax[bh] = red[0];
}

// ---------------------------------------------------------------------------
// Kernel 3: fused softmax + (alpha @ z) + head-mean + bias, flash-style.
// Block = 128 threads; wave w owns dest rows [i0+16w, +16). alpha values are
// generated directly in the WMMA A-fragment layout (one row per lane).
// z chunks are staged global->LDS with async DMA, double-buffered so the DMA
// of chunk k+1 overlaps the 8 WMMAs of chunk k.
// ---------------------------------------------------------------------------
__global__ __launch_bounds__(128) void gat_attend(
    const _Float16* __restrict__ zT,     // [B,H,C,N]
    const float* __restrict__ s_src,     // [B,H,N]
    const float* __restrict__ s_dst,     // [B,H,N]
    const float* __restrict__ smaxArr,   // [B,H]
    const float* __restrict__ bias,      // [C]
    float* __restrict__ out)             // [B,C,N]
{
  __shared__ __align__(16) _Float16 ldsB[2][Cc * PAD];  // double-buffered Bt[c][j]
  __shared__ __align__(16) float ss[Nn];                // s_src for this (b,h)
  __shared__ float wsum[4 * 16];

  const int b    = blockIdx.x;
  const int i0   = blockIdx.y * 64;
  const int t    = threadIdx.x;
  const int wave = t >> 5;
  const int lane = t & 31;
  const int lrow = lane & 15;
  const int hiH  = lane >> 4;
  const int iRow = i0 + wave * 16 + lrow;  // A-matrix row held by this lane
  const int kbase = hiH * 8;

  float outAcc[8][8];
#pragma unroll
  for (int tc = 0; tc < 8; ++tc)
#pragma unroll
    for (int r = 0; r < 8; ++r) outAcc[tc][r] = 0.f;

  for (int h = 0; h < Hh; ++h) {
    const int bh = b * Hh + h;
    const float* ssrc = s_src + (size_t)bh * Nn;
    const _Float16* zrow = zT + ((size_t)bh * Cc + t) * Nn;  // this thread's c row

    // Stage s_src vector + chunk 0 of z (async DMA when available).
#if HAVE_ASYNC
    for (int i = t; i < Nn; i += 128) ASYNC_B32(ssrc + i, ss + i);
#pragma unroll
    for (int q = 0; q < 4; ++q)
      ASYNC_B128(zrow + 8 * q, &ldsB[0][t * PAD] + 8 * q);
    ASYNC_WAIT();
#else
    for (int i = t; i < Nn; i += 128) ss[i] = ssrc[i];
    {
      const v8h* src = (const v8h*)zrow;
      v8h* dst = (v8h*)&ldsB[0][t * PAD];
      dst[0] = src[0]; dst[1] = src[1]; dst[2] = src[2]; dst[3] = src[3];
    }
#endif
    __syncthreads();

    const float d   = s_dst[(size_t)bh * Nn + iRow];
    const float dm  = d + smaxArr[bh];
    const float mrow = fmaxf(dm, NEG * dm);  // exact row max (lrelu monotone)

    v8f acc[8];
#pragma unroll
    for (int i = 0; i < 8; ++i) acc[i] = zero8();
    float rsum = 0.f;

    int cur = 0;
    for (int j0 = 0; j0 < Nn; j0 += 32) {
      const bool more = (j0 + 32) < Nn;  // uniform
      // Kick off DMA for the next chunk into the other buffer.
      if (more) {
        const _Float16* g = zrow + j0 + 32;
        _Float16* l = &ldsB[cur ^ 1][t * PAD];
#if HAVE_ASYNC
#pragma unroll
        for (int q = 0; q < 4; ++q) ASYNC_B128(g + 8 * q, l + 8 * q);
#else
        const v8h* src = (const v8h*)g;
        v8h* dst = (v8h*)l;
        dst[0] = src[0]; dst[1] = src[1]; dst[2] = src[2]; dst[3] = src[3];
#endif
      }

      // A fragment = exp(lrelu(d_i + s_j) - mrow), built in-register.
      const float* sp = ss + j0 + kbase;
      v16h afrag;
#pragma unroll
      for (int q = 0; q < 8; ++q) {
        const float e0 = d + sp[q];
        const float p0 = __expf(fmaxf(e0, NEG * e0) - mrow);
        const float e1 = d + sp[16 + q];
        const float p1 = __expf(fmaxf(e1, NEG * e1) - mrow);
        rsum += p0 + p1;
        afrag[q]     = (_Float16)p0;
        afrag[8 + q] = (_Float16)p1;
      }
      // 8 WMMAs on the current buffer overlap the in-flight DMA above.
      const _Float16* base = ldsB[cur];
#pragma unroll
      for (int tc = 0; tc < 8; ++tc) {
        const _Float16* brow = base + (tc * 16 + lrow) * PAD + hiH * 8;
        v16h bfrag = make16(*(const v8h*)brow, *(const v8h*)(brow + 16));
        acc[tc] = __builtin_amdgcn_wmma_f32_16x16x32_f16(
            false, afrag, false, bfrag, (short)0, acc[tc], false, false);
      }

      if (more) ASYNC_WAIT();  // next buffer landed
      __syncthreads();
      cur ^= 1;
    }

    // Row-sum: row i held by lanes (lrow) and (lrow+16).
    rsum += __shfl_xor(rsum, 16, 32);
    if (lane < 16) wsum[wave * 16 + lrow] = rsum;
    __syncthreads();
    float inv[8];
#pragma unroll
    for (int r = 0; r < 8; ++r)
      inv[r] = 0.5f / wsum[wave * 16 + hiH * 8 + r];  // 0.5 = head mean
#pragma unroll
    for (int tc = 0; tc < 8; ++tc)
#pragma unroll
      for (int r = 0; r < 8; ++r) outAcc[tc][r] += acc[tc][r] * inv[r];
    __syncthreads();
  }

  // Epilogue: out[b, c, n] = acc + bias[c]; 8 consecutive n per lane/tile.
#pragma unroll
  for (int tc = 0; tc < 8; ++tc) {
    const int c = tc * 16 + lrow;
    const float bv = bias[c];
    float* op = out + ((size_t)b * Cc + c) * Nn + i0 + wave * 16 + hiH * 8;
    v4f lo = {outAcc[tc][0] + bv, outAcc[tc][1] + bv,
              outAcc[tc][2] + bv, outAcc[tc][3] + bv};
    v4f hi = {outAcc[tc][4] + bv, outAcc[tc][5] + bv,
              outAcc[tc][6] + bv, outAcc[tc][7] + bv};
    *(v4f*)op = lo;
    *(v4f*)(op + 4) = hi;
  }
}

// ---------------------------------------------------------------------------
extern "C" void kernel_launch(void* const* d_in, const int* in_sizes, int n_in,
                              void* d_out, int out_size, void* d_ws, size_t ws_size,
                              hipStream_t stream) {
  (void)in_sizes; (void)n_in; (void)out_size; (void)ws_size;
  const float* x       = (const float*)d_in[0];  // [B,S,N]
  const float* Wm      = (const float*)d_in[1];  // [H,S,C]
  const float* att_src = (const float*)d_in[2];  // [H,C]
  const float* att_dst = (const float*)d_in[3];  // [H,C]
  const float* bias    = (const float*)d_in[4];  // [C]
  float* out = (float*)d_out;                    // [B,C,N]

  char* ws = (char*)d_ws;
  _Float16* zT = (_Float16*)ws;  // B*H*C*N f16 = 16.78 MB
  const size_t zbytes = (size_t)Bsz * Hh * Cc * Nn * sizeof(_Float16);
  float* s_src = (float*)(ws + zbytes);          // B*H*N f32
  float* s_dst = s_src + (size_t)Bsz * Hh * Nn;  // B*H*N f32
  float* smax  = s_dst + (size_t)Bsz * Hh * Nn;  // B*H f32

  dim3 g1(Bsz * Hh, Nn / 64);
  gat_project<<<g1, 128, 0, stream>>>(x, Wm, att_src, att_dst, zT, s_src, s_dst);
  gat_smax<<<Bsz * Hh, 128, 0, stream>>>(s_src, smax);
  dim3 g2(Bsz, Nn / 64);
  gat_attend<<<g2, 128, 0, stream>>>(zT, s_src, s_dst, smax, bias, out);
}
